// CausalAttention_6176162971744
// MI455X (gfx1250) — matrix operand order repaired and checked by reading, not compiled, not run on hardware
//
#include <hip/hip_runtime.h>

// ---------------- problem constants ----------------
#define D_MODEL 1024
#define N_HEADS 16
#define DHEAD   64
#define TMAX    2048
#define BATCH   4
#define BT      (BATCH * TMAX)   // 8192 rows
#define QKV3    (3 * D_MODEL)    // 3072

typedef _Float16 v8h  __attribute__((ext_vector_type(8)));
typedef _Float16 v16h __attribute__((ext_vector_type(16)));
typedef float    v8f  __attribute__((ext_vector_type(8)));

// ---------------- helpers ----------------
static __device__ __forceinline__ v16h cat8(v8h lo, v8h hi) {
    v16h r;
#pragma unroll
    for (int i = 0; i < 8; ++i) { r[i] = lo[i]; r[i + 8] = hi[i]; }
    return r;
}

static __device__ __forceinline__ v8f vzero8() {
    v8f z;
#pragma unroll
    for (int i = 0; i < 8; ++i) z[i] = 0.0f;
    return z;
}

static __device__ __forceinline__ v8f wmma16(v16h a, v16h b, v8f c) {
    // D = A(16x32 f16) * B(32x16 f16) + C(16x16 f32)
    return __builtin_amdgcn_wmma_f32_16x16x32_f16(
        /*neg_a=*/false, a, /*neg_b=*/false, b,
        /*c_mod=*/(short)0, c, /*reuse_a=*/false, /*reuse_b=*/false);
}

// reductions across the 16-lane half of a wave32 (xor masks < 16 stay in-half)
static __device__ __forceinline__ float redmax16(float v) {
#pragma unroll
    for (int m = 1; m < 16; m <<= 1) v = fmaxf(v, __shfl_xor(v, m, 32));
    return v;
}
static __device__ __forceinline__ float redsum16(float v) {
#pragma unroll
    for (int m = 1; m < 16; m <<= 1) v += __shfl_xor(v, m, 32);
    return v;
}

// ---------------- precision conversion kernels ----------------
__global__ void f32_to_f16_kernel(const float* __restrict__ src,
                                  _Float16* __restrict__ dst, int n) {
    int i = blockIdx.x * blockDim.x + threadIdx.x;
    if (i < n) dst[i] = (_Float16)src[i];
}

// src[R][C] f32 row-major  ->  dst[C][R] f16 (transposed)
__global__ void transpose_f32_to_f16_kernel(const float* __restrict__ src,
                                            _Float16* __restrict__ dst,
                                            int R, int C) {
    int i = blockIdx.x * blockDim.x + threadIdx.x;
    if (i < R * C) {
        int r = i / C, c = i % C;
        dst[(size_t)c * R + r] = (_Float16)src[i];
    }
}

// ---------------- GEMM 1: qkv = x @ W_qkv + b, scattered to Q/K/Vt ----------------
// Each wave computes a 32(M) x 64(N) tile: two A fragments share every B fragment,
// so the steady state is ~1 b128 load per WMMA (2x the compute intensity of 16x64).
// xh:   [BT][D_MODEL] f16 (row major)
// wt:   [QKV3][D_MODEL] f16 (W_qkv transposed)
// Qb,Kb: [B][H][T][DHEAD] f16 ; Vt: [B][H][DHEAD][T] f16
__global__ __launch_bounds__(256) void gemm_qkv_kernel(
    const _Float16* __restrict__ xh, const _Float16* __restrict__ wt,
    const float* __restrict__ bias,
    _Float16* __restrict__ Qb, _Float16* __restrict__ Kb,
    _Float16* __restrict__ Vt) {
    const int w    = threadIdx.x >> 5;
    const int lane = threadIdx.x & 31;
    const int wid  = blockIdx.x * 8 + w;
    const int nt   = wid % (QKV3 / 64);   // 48 n-tiles of 64
    const int mt   = wid / (QKV3 / 64);   // 256 m-tiles of 32
    const int m0 = mt * 32, n0 = nt * 64;
    const int col = lane & 15;
    const int ho  = (lane < 16) ? 0 : 8;
    const int aoff0 = (lane < 16) ? 0 : 16;
    const int aoff1 = (lane < 16) ? 8 : 24;
    const int boff  = (lane < 16) ? 0 : 16;

    v8f acc[2][4];
#pragma unroll
    for (int s = 0; s < 2; ++s)
#pragma unroll
        for (int j = 0; j < 4; ++j) acc[s][j] = vzero8();

    const _Float16* ar0 = xh + (size_t)(m0 + col) * D_MODEL;
    const _Float16* ar1 = xh + (size_t)(m0 + 16 + col) * D_MODEL;

    for (int kc = 0; kc < D_MODEL; kc += 32) {
        v16h a0 = cat8(*(const v8h*)(ar0 + kc + aoff0),
                       *(const v8h*)(ar0 + kc + aoff1));
        v16h a1 = cat8(*(const v8h*)(ar1 + kc + aoff0),
                       *(const v8h*)(ar1 + kc + aoff1));
#pragma unroll
        for (int j = 0; j < 4; ++j) {
            const _Float16* br = wt + (size_t)(n0 + j * 16 + col) * D_MODEL
                                 + kc + boff;
            v16h b = *(const v16h*)br;
            acc[0][j] = wmma16(a0, b, acc[0][j]);
            acc[1][j] = wmma16(a1, b, acc[1][j]);
        }
    }

#pragma unroll
    for (int s = 0; s < 2; ++s) {
#pragma unroll
        for (int j = 0; j < 4; ++j) {
            const int n = n0 + j * 16 + col;
            const float bs = bias[n];
#pragma unroll
            for (int i = 0; i < 8; ++i) {
                const int row = m0 + s * 16 + i + ho;
                const int bb = row >> 11;      // / TMAX
                const int t  = row & (TMAX - 1);
                const _Float16 hv = (_Float16)(acc[s][j][i] + bs);
                if (n < D_MODEL) {
                    const int h = n >> 6, d = n & 63;
                    Qb[(((size_t)(bb * N_HEADS + h) * TMAX + t) << 6) + d] = hv;
                } else if (n < 2 * D_MODEL) {
                    const int nn = n - D_MODEL, h = nn >> 6, d = nn & 63;
                    Kb[(((size_t)(bb * N_HEADS + h) * TMAX + t) << 6) + d] = hv;
                } else {
                    const int nn = n - 2 * D_MODEL, h = nn >> 6, d = nn & 63;
                    Vt[(((size_t)(bb * N_HEADS + h) * DHEAD + d) << 11) + t] = hv;
                }
            }
        }
    }
}

// ---------------- flash-style causal attention ----------------
// one wave per (b,h, 16-row q tile); key blocks of 32
__global__ __launch_bounds__(256) void attn_kernel(
    const _Float16* __restrict__ Qb, const _Float16* __restrict__ Kb,
    const _Float16* __restrict__ Vt, _Float16* __restrict__ ao) {
    __shared__ __align__(32) _Float16 plds[8][16 * 32];  // 1 KB per wave

    const int w    = threadIdx.x >> 5;
    const int lane = threadIdx.x & 31;
    const int wid  = blockIdx.x * 8 + w;
    const int qt = wid & (TMAX / 16 - 1);  // 128 q tiles
    const int bh = wid >> 7;               // 64 (b,h) pairs
    const int q0 = qt << 4;

    const _Float16* Qp = Qb + (size_t)bh * TMAX * DHEAD;
    const _Float16* Kp = Kb + (size_t)bh * TMAX * DHEAD;
    const _Float16* Vp = Vt + (size_t)bh * DHEAD * TMAX;

    const int col = lane & 15;
    const int ho  = (lane < 16) ? 0 : 8;
    const int aoff0 = (lane < 16) ? 0 : 16;
    const int aoff1 = (lane < 16) ? 8 : 24;
    const int boff  = (lane < 16) ? 0 : 16;

    // Q A-fragments for the two 32-wide k-chunks of d_head=64
    const _Float16* qrow = Qp + (size_t)(q0 + col) * DHEAD;
    v16h aQ[2];
#pragma unroll
    for (int c = 0; c < 2; ++c) {
        aQ[c] = cat8(*(const v8h*)(qrow + c * 32 + aoff0),
                     *(const v8h*)(qrow + c * 32 + aoff1));
    }

    float mrow[8], lrow[8];
    v8f acc[4];
#pragma unroll
    for (int i = 0; i < 8; ++i) { mrow[i] = -1e30f; lrow[i] = 0.0f; }
#pragma unroll
    for (int j = 0; j < 4; ++j) acc[j] = vzero8();

    const float scale = 0.125f;  // 1/sqrt(64)
    const int nkb = (q0 + 16 + 31) >> 5;  // key blocks covering keys <= q0+15

    for (int kb = 0; kb < nkb; ++kb) {
        v8f S0 = vzero8(), S1 = vzero8();
#pragma unroll
        for (int c = 0; c < 2; ++c) {
            const _Float16* k0 = Kp + (size_t)(kb * 32 + col) * DHEAD
                                 + c * 32 + boff;
            const _Float16* k1 = Kp + (size_t)(kb * 32 + 16 + col) * DHEAD
                                 + c * 32 + boff;
            S0 = wmma16(aQ[c], *(const v16h*)k0, S0);
            S1 = wmma16(aQ[c], *(const v16h*)k1, S1);
        }

        // causal mask + online softmax (row lives across the 16-lane half)
        float p0[8], p1[8];
        const int c0 = kb * 32 + col, c1 = c0 + 16;
#pragma unroll
        for (int i = 0; i < 8; ++i) {
            const int row = q0 + i + ho;
            const float s0 = (c0 <= row) ? S0[i] * scale : -1e30f;
            const float s1 = (c1 <= row) ? S1[i] * scale : -1e30f;
            const float bm = redmax16(fmaxf(s0, s1));
            const float mn = fmaxf(mrow[i], bm);
            const float al = __expf(mrow[i] - mn);
            const float e0 = __expf(s0 - mn);
            const float e1 = __expf(s1 - mn);
            lrow[i] = lrow[i] * al + redsum16(e0 + e1);
            mrow[i] = mn;
            p0[i] = e0; p1[i] = e1;
#pragma unroll
            for (int j = 0; j < 4; ++j) acc[j][i] *= al;
        }

        // re-layout P from C/D format to A format via LDS (same-wave DS is in-order)
        asm volatile("" ::: "memory");
#pragma unroll
        for (int i = 0; i < 8; ++i) {
            plds[w][(i + ho) * 32 + col]      = (_Float16)p0[i];
            plds[w][(i + ho) * 32 + 16 + col] = (_Float16)p1[i];
        }
        asm volatile("s_wait_dscnt 0" ::: "memory");
        const _Float16* pr = &plds[w][col * 32];
        v16h aP = cat8(*(const v8h*)(pr + aoff0), *(const v8h*)(pr + aoff1));

        // out += P(16x32) * V(32x64), Vt gives contiguous keys per d-column
#pragma unroll
        for (int j = 0; j < 4; ++j) {
            const _Float16* vp = Vp + (size_t)(j * 16 + col) * TMAX
                                 + kb * 32 + boff;
            acc[j] = wmma16(aP, *(const v16h*)vp, acc[j]);
        }
    }

    // normalize + write merged-head layout ao[b][t][h*64+d] (f16)
    const int b = bh >> 4, h = bh & 15;
#pragma unroll
    for (int j = 0; j < 4; ++j) {
#pragma unroll
        for (int i = 0; i < 8; ++i) {
            const int t = q0 + i + ho;
            const float v = acc[j][i] / lrow[i];
            ao[((size_t)(b * TMAX + t)) * D_MODEL + h * DHEAD + j * 16 + col] =
                (_Float16)v;
        }
    }
}

// ---------------- GEMM 2: out = ao @ W_proj + b (f32 out) ----------------
// 32(M) x 64(N) wave tiles, B fragments shared across both M sub-tiles.
__global__ __launch_bounds__(256) void gemm_proj_kernel(
    const _Float16* __restrict__ A, const _Float16* __restrict__ Wt,
    const float* __restrict__ bias, float* __restrict__ out) {
    const int w    = threadIdx.x >> 5;
    const int lane = threadIdx.x & 31;
    const int wid  = blockIdx.x * 8 + w;
    const int nt = wid & 15;   // 16 n-tiles of 64
    const int mt = wid >> 4;   // 256 m-tiles of 32
    const int m0 = mt * 32, n0 = nt * 64;
    const int col = lane & 15;
    const int ho  = (lane < 16) ? 0 : 8;
    const int aoff0 = (lane < 16) ? 0 : 16;
    const int aoff1 = (lane < 16) ? 8 : 24;
    const int boff  = (lane < 16) ? 0 : 16;

    v8f acc[2][4];
#pragma unroll
    for (int s = 0; s < 2; ++s)
#pragma unroll
        for (int j = 0; j < 4; ++j) acc[s][j] = vzero8();

    const _Float16* ar0 = A + (size_t)(m0 + col) * D_MODEL;
    const _Float16* ar1 = A + (size_t)(m0 + 16 + col) * D_MODEL;

    for (int kc = 0; kc < D_MODEL; kc += 32) {
        v16h a0 = cat8(*(const v8h*)(ar0 + kc + aoff0),
                       *(const v8h*)(ar0 + kc + aoff1));
        v16h a1 = cat8(*(const v8h*)(ar1 + kc + aoff0),
                       *(const v8h*)(ar1 + kc + aoff1));
#pragma unroll
        for (int j = 0; j < 4; ++j) {
            const _Float16* br = Wt + (size_t)(n0 + j * 16 + col) * D_MODEL
                                 + kc + boff;
            v16h b = *(const v16h*)br;
            acc[0][j] = wmma16(a0, b, acc[0][j]);
            acc[1][j] = wmma16(a1, b, acc[1][j]);
        }
    }

#pragma unroll
    for (int s = 0; s < 2; ++s) {
#pragma unroll
        for (int j = 0; j < 4; ++j) {
            const int n = n0 + j * 16 + col;
            const float bs = bias[n];
#pragma unroll
            for (int i = 0; i < 8; ++i)
                out[(size_t)(m0 + s * 16 + i + ho) * D_MODEL + n] =
                    acc[s][j][i] + bs;
        }
    }
}

// ---------------- launch ----------------
extern "C" void kernel_launch(void* const* d_in, const int* in_sizes, int n_in,
                              void* d_out, int out_size, void* d_ws, size_t ws_size,
                              hipStream_t stream) {
    (void)in_sizes; (void)n_in; (void)out_size; (void)ws_size;
    const float* x      = (const float*)d_in[0];  // [B,T,C]
    const float* W_qkv  = (const float*)d_in[1];  // [C,3C]
    const float* b_qkv  = (const float*)d_in[2];  // [3C]
    const float* W_proj = (const float*)d_in[3];  // [C,C]
    const float* b_proj = (const float*)d_in[4];  // [C]
    float* out = (float*)d_out;                   // [B,T,C]

    // workspace layout (bytes)
    char* ws = (char*)d_ws;
    size_t off = 0;
    _Float16* xh     = (_Float16*)(ws + off); off += (size_t)BT * D_MODEL * 2;      // 16 MB
    _Float16* wtqkv  = (_Float16*)(ws + off); off += (size_t)QKV3 * D_MODEL * 2;    //  6 MB
    _Float16* wtproj = (_Float16*)(ws + off); off += (size_t)D_MODEL * D_MODEL * 2; //  2 MB
    _Float16* Qb     = (_Float16*)(ws + off); off += (size_t)BT * D_MODEL * 2;      // 16 MB
    _Float16* Kb     = (_Float16*)(ws + off); off += (size_t)BT * D_MODEL * 2;      // 16 MB
    _Float16* Vt     = (_Float16*)(ws + off); off += (size_t)BT * D_MODEL * 2;      // 16 MB
    _Float16* ao     = (_Float16*)(ws + off); off += (size_t)BT * D_MODEL * 2;      // 16 MB

    // 1) precision conversion / weight transposition
    {
        int n = BT * D_MODEL;
        f32_to_f16_kernel<<<(n + 255) / 256, 256, 0, stream>>>(x, xh, n);
    }
    {
        int n = D_MODEL * QKV3;
        transpose_f32_to_f16_kernel<<<(n + 255) / 256, 256, 0, stream>>>(
            W_qkv, wtqkv, D_MODEL, QKV3);
    }
    {
        int n = D_MODEL * D_MODEL;
        transpose_f32_to_f16_kernel<<<(n + 255) / 256, 256, 0, stream>>>(
            W_proj, wtproj, D_MODEL, D_MODEL);
    }

    // 2) qkv GEMM: (BT/32)*(3C/64) wave tiles, 8 waves/block
    {
        int waves = (BT / 32) * (QKV3 / 64);  // 12288
        gemm_qkv_kernel<<<waves / 8, 256, 0, stream>>>(xh, wtqkv, b_qkv, Qb, Kb, Vt);
    }

    // 3) causal attention: B*H*(T/16) wave tiles
    {
        int waves = BATCH * N_HEADS * (TMAX / 16);  // 8192
        attn_kernel<<<waves / 8, 256, 0, stream>>>(Qb, Kb, Vt, ao);
    }

    // 4) projection GEMM: (BT/32)*(C/64) wave tiles
    {
        int waves = (BT / 32) * (D_MODEL / 64);  // 4096
        gemm_proj_kernel<<<waves / 8, 256, 0, stream>>>(ao, wtproj, b_proj, out);
    }
}
